// MoEMLP_26164940767786
// MI455X (gfx1250) — matrix operand order repaired and checked by reading, not compiled, not run on hardware
//
#include <hip/hip_runtime.h>
#include <hip/hip_bf16.h>
#include <math.h>

// ---------------------------------------------------------------------------
// MoE MLP forward for MI455X (gfx1250, wave32, WMMA).
// N=8192 tokens, D=1024 model dim, H=4096 hidden, E=8 experts, top-2,
// capacity C = 2560. Expert GEMMs: v_wmma_f32_16x16x32_bf16, double-buffered
// LDS with async global->LDS staging of the bf16 A operand (ASYNCcnt).
// ---------------------------------------------------------------------------

typedef __attribute__((ext_vector_type(16))) __bf16        v16bf;
typedef __attribute__((ext_vector_type(8)))  float         v8f;
typedef __attribute__((ext_vector_type(8)))  unsigned int  v8u;
typedef __attribute__((ext_vector_type(4)))  unsigned int  u32x4;
typedef __attribute__((ext_vector_type(4)))  float         f32x4;
typedef int gcc_i32x4 __attribute__((vector_size(16)));    // matches builtin's V4i

constexpr int Ntok = 8192;
constexpr int Dm   = 1024;
constexpr int Hm   = 4096;
constexpr int Ex   = 8;
constexpr int Cap  = 2560;   // int(2 * 1.25 * 8192 / 8)

#if __has_builtin(__builtin_amdgcn_global_load_async_to_lds_b128)
#define USE_ASYNC_LDS 1
#else
#define USE_ASYNC_LDS 0
#endif

__device__ __forceinline__ unsigned short f2bf(float f) {
  unsigned int u = __builtin_bit_cast(unsigned int, f);
  u += 0x7fffu + ((u >> 16) & 1u);          // round-to-nearest-even
  return (unsigned short)(u >> 16);
}

#if USE_ASYNC_LDS
// 16B global -> LDS async copy (ASYNCcnt). OFF applies to both addresses.
// Builtin signature: (v4i addrspace(1)*, v4i addrspace(3)*, imm offset, imm cpol).
template <int OFF>
__device__ __forceinline__ void async16(const unsigned short* g, unsigned int* l) {
  __builtin_amdgcn_global_load_async_to_lds_b128(
      (__attribute__((address_space(1))) gcc_i32x4*)(unsigned long long)g,
      (__attribute__((address_space(3))) gcc_i32x4*)(unsigned int)(unsigned long long)l,
      OFF, 0);
}
#endif

__device__ __forceinline__ void wait_async0() {
#if USE_ASYNC_LDS
#if __has_builtin(__builtin_amdgcn_s_wait_asynccnt)
  __builtin_amdgcn_s_wait_asynccnt(0);
#else
  asm volatile("s_wait_asynccnt 0x0" ::: "memory");
#endif
#endif
}

// ---------------------------------------------------------------------------
// Gating: one wave per token. logits = x[t] @ wg, softmax, top-2.
// ---------------------------------------------------------------------------
__global__ void __launch_bounds__(256) k_gate(
    const float* __restrict__ x, const float* __restrict__ wg,
    float* __restrict__ scores,
    int* __restrict__ eid0, int* __restrict__ eid1,
    float* __restrict__ gv0, float* __restrict__ gv1,
    int* __restrict__ counts0)
{
  __shared__ float swg[Dm * Ex];            // 32 KB
  const int tid = threadIdx.x;
  for (int i = tid * 4; i < Dm * Ex; i += 256 * 4)
    *(f32x4*)&swg[i] = *(const f32x4*)&wg[i];
  __syncthreads();

  const int lane = tid & 31;
  const int wv   = tid >> 5;
  const int t    = blockIdx.x * 8 + wv;

  float acc[Ex];
#pragma unroll
  for (int e = 0; e < Ex; ++e) acc[e] = 0.f;

  for (int k = lane; k < Dm; k += 32) {
    const float xv = x[(size_t)t * Dm + k];
#pragma unroll
    for (int e = 0; e < Ex; ++e) acc[e] += xv * swg[k * Ex + e];
  }
#pragma unroll
  for (int e = 0; e < Ex; ++e) {
#pragma unroll
    for (int off = 16; off > 0; off >>= 1)
      acc[e] += __shfl_xor(acc[e], off, 32);
  }

  if (lane == 0) {
    float m = acc[0];
#pragma unroll
    for (int e = 1; e < Ex; ++e) m = fmaxf(m, acc[e]);
    float p[Ex], s = 0.f;
#pragma unroll
    for (int e = 0; e < Ex; ++e) { p[e] = expf(acc[e] - m); s += p[e]; }
    const float inv = 1.f / s;
#pragma unroll
    for (int e = 0; e < Ex; ++e) { p[e] *= inv; scores[(size_t)t * Ex + e] = p[e]; }

    int b0 = 0; float g0 = p[0];
#pragma unroll
    for (int e = 1; e < Ex; ++e) if (p[e] > g0) { g0 = p[e]; b0 = e; }  // ties -> low idx
    int b1 = -1; float g1 = -1.f;
#pragma unroll
    for (int e = 0; e < Ex; ++e) if (e != b0 && p[e] > g1) { g1 = p[e]; b1 = e; }

    eid0[t] = b0; gv0[t] = g0;
    eid1[t] = b1; gv1[t] = g1;
    atomicAdd(&counts0[b0], 1);             // integer -> deterministic
  }
}

// ---------------------------------------------------------------------------
// l_aux = E * sum_e mean(scores[:,e]) * (counts0[e]/N), deterministic tree.
// ---------------------------------------------------------------------------
__global__ void __launch_bounds__(256) k_laux(
    const float* __restrict__ scores, const int* __restrict__ counts0,
    float* __restrict__ out_laux)
{
  __shared__ float red[256];
  const int tid = threadIdx.x;
  float pm[Ex];
#pragma unroll
  for (int e = 0; e < Ex; ++e) pm[e] = 0.f;
  for (int i = tid; i < Ntok; i += 256) {
#pragma unroll
    for (int e = 0; e < Ex; ++e) pm[e] += scores[(size_t)i * Ex + e];
  }
  float me[Ex];
#pragma unroll
  for (int e = 0; e < Ex; ++e) {
    red[tid] = pm[e];
    __syncthreads();
    for (int s = 128; s > 0; s >>= 1) {
      if (tid < s) red[tid] += red[tid + s];
      __syncthreads();
    }
    if (tid == 0) me[e] = red[0];
    __syncthreads();
  }
  if (tid == 0) {
    float l = 0.f;
#pragma unroll
    for (int e = 0; e < Ex; ++e)
      l += (me[e] / (float)Ntok) * ((float)counts0[e] / (float)Ntok);
    *out_laux = l * (float)Ex;
  }
}

// ---------------------------------------------------------------------------
// Capacity ranks, one WAVE per token (32-way parallel scan + shfl reduce).
// pos_j(t) = #{t' : e'=e && (g' > g || (g'==g && t' < t))}  == stable
// argsort(-g) + per-expert cumsum rank of the reference.
// ---------------------------------------------------------------------------
__global__ void __launch_bounds__(256) k_rank(
    const int* __restrict__ eid0, const int* __restrict__ eid1,
    const float* __restrict__ gv0, const float* __restrict__ gv1,
    const int* __restrict__ counts0,
    int* __restrict__ flat0, int* __restrict__ flat1,
    float* __restrict__ w0, float* __restrict__ w1)
{
  __shared__ int   se0[2048], se1[2048];
  __shared__ float sg0[2048], sg1[2048];   // 32 KB total
  const int tid  = threadIdx.x;
  const int lane = tid & 31;
  const int wv   = tid >> 5;
  const int t    = blockIdx.x * 8 + wv;    // token handled by this wave
  const int   me0 = eid0[t], me1 = eid1[t];
  const float mg0 = gv0[t],  mg1 = gv1[t];
  int p0 = 0, p1 = 0;

  for (int base = 0; base < Ntok; base += 2048) {
    __syncthreads();
    for (int i = tid; i < 2048; i += 256) {
      se0[i] = eid0[base + i]; sg0[i] = gv0[base + i];
      se1[i] = eid1[base + i]; sg1[i] = gv1[base + i];
    }
    __syncthreads();
    for (int i = lane; i < 2048; i += 32) {
      const int u = base + i;
      p0 += (se0[i] == me0 && (sg0[i] > mg0 || (sg0[i] == mg0 && u < t))) ? 1 : 0;
      p1 += (se1[i] == me1 && (sg1[i] > mg1 || (sg1[i] == mg1 && u < t))) ? 1 : 0;
    }
  }
#pragma unroll
  for (int off = 16; off > 0; off >>= 1) {
    p0 += __shfl_xor(p0, off, 32);
    p1 += __shfl_xor(p1, off, 32);
  }
  if (lane == 0) {
    p1 += counts0[me1];
    flat0[t] = (p0 < Cap) ? me0 * Cap + p0 : -1;
    w0[t]    = (p0 < Cap) ? mg0 : 0.f;
    flat1[t] = (p1 < Cap) ? me1 * Cap + p1 : -1;
    w1[t]    = (p1 < Cap) ? mg1 : 0.f;
  }
}

// ---------------------------------------------------------------------------
// Scatter x -> disp (bf16). Slots are collision-free across the two lists.
// ---------------------------------------------------------------------------
__global__ void __launch_bounds__(256) k_scatter(
    const float* __restrict__ x,
    const int* __restrict__ flat0, const int* __restrict__ flat1,
    unsigned short* __restrict__ disp)
{
  const int t = blockIdx.x, tid = threadIdx.x;
  const int d = tid * 4;
  const f32x4 xv = *(const f32x4*)&x[(size_t)t * Dm + d];
  unsigned int lo = (unsigned)f2bf(xv.x) | ((unsigned)f2bf(xv.y) << 16);
  unsigned int hi = (unsigned)f2bf(xv.z) | ((unsigned)f2bf(xv.w) << 16);
  const int f0 = flat0[t];
  if (f0 >= 0) {
    unsigned int* p = (unsigned int*)&disp[(size_t)f0 * Dm + d];
    p[0] = lo; p[1] = hi;
  }
  const int f1 = flat1[t];
  if (f1 >= 0) {
    unsigned int* p = (unsigned int*)&disp[(size_t)f1 * Dm + d];
    p[0] = lo; p[1] = hi;
  }
}

// ---------------------------------------------------------------------------
// Batched expert GEMM, double-buffered 128x128x32 tiles, 8 waves/block.
// A (MxK bf16): async global->LDS (row-major tile, row stride 64B).
// B (KxN f32): global->regs, f32->bf16 pair-pack, LDS col-major (col stride 64B).
// Fragments: two ds_load_b128 each, A and B in the same contiguous K order
// (lanes 0-15 K=0..15, lanes 16-31 K=16..31).
// GELU_BF16=1: +bias, exact-erf GELU, bf16 out.  =0: +bias, f32 out.
// ---------------------------------------------------------------------------
template <int GELU_BF16>
__global__ void __launch_bounds__(256) k_gemm(
    const unsigned short* __restrict__ Abase, const float* __restrict__ Bbase,
    const float* __restrict__ biasBase, void* __restrict__ Obase,
    int M, int Nn, int K, int mTiles,
    long long aBatch, long long bBatch, long long biasBatch, long long oBatch)
{
  __shared__ unsigned int sA[2][128 * 16];  // 2 x 8 KB
  __shared__ unsigned int sB[2][128 * 16];  // 2 x 8 KB

  const int e = blockIdx.y;
  const unsigned short* A   = Abase    + (size_t)e * aBatch;
  const float*          B   = Bbase    + (size_t)e * bBatch;
  const float*          bia = biasBase + (size_t)e * biasBatch;

  const int tid  = threadIdx.x;
  const int m0   = (blockIdx.x % mTiles) * 128;
  const int n0   = (blockIdx.x / mTiles) * 128;
  const int lane = tid & 31, wv = tid >> 5;
  const int wm   = (wv & 1) * 64, wn = (wv >> 1) * 32;
  const int l15  = lane & 15, lhi = lane >> 4;

  v8f acc[4][2];
#pragma unroll
  for (int i = 0; i < 4; ++i)
#pragma unroll
    for (int j = 0; j < 2; ++j) acc[i][j] = v8f{0.f,0.f,0.f,0.f,0.f,0.f,0.f,0.f};

  // A staging: 32B per thread.  B staging: 2 K-rows x 8 cols per thread.
  const int arow = tid >> 1, ahalf = tid & 1;
  const int bkp  = tid >> 4, bc0 = (tid & 15) * 8;
  const unsigned short* agp = A + (size_t)(m0 + arow) * K + ahalf * 16;
  unsigned int* aldp = &sA[0][0] + (arow * 16 + ahalf * 8);   // buf stride 128*16

  float br[16];

  auto stage_a = [&](int k0, int buf) {
#if USE_ASYNC_LDS
    async16<0>(agp + k0, aldp + buf * 128 * 16);
    async16<16>(agp + k0, aldp + buf * 128 * 16);
#else
    const u32x4* g = (const u32x4*)(agp + k0);
    u32x4 q0 = g[0], q1 = g[1];
    u32x4* s = (u32x4*)(aldp + buf * 128 * 16);
    s[0] = q0; s[1] = q1;
#endif
  };
  auto load_b = [&](int k0) {
    const float* g0p = B + (size_t)(k0 + 2 * bkp) * Nn + n0 + bc0;
    const float* g1p = g0p + Nn;
    *(f32x4*)&br[0]  = *(const f32x4*)&g0p[0];
    *(f32x4*)&br[4]  = *(const f32x4*)&g0p[4];
    *(f32x4*)&br[8]  = *(const f32x4*)&g1p[0];
    *(f32x4*)&br[12] = *(const f32x4*)&g1p[4];
  };
  auto store_b = [&](int buf) {
#pragma unroll
    for (int i = 0; i < 8; ++i) {
      unsigned int dw = (unsigned)f2bf(br[i]) | ((unsigned)f2bf(br[8 + i]) << 16);
      sB[buf][(bc0 + i) * 16 + bkp] = dw;   // col-major: K pair (2bkp,2bkp+1)
    }
  };
  auto compute = [&](int buf) {
    v16bf af[4], bfg[2];
#pragma unroll
    for (int i = 0; i < 4; ++i) {
      const int idx = (wm + i * 16 + l15) * 16 + lhi * 8;  // bytes: row*64+lhi*32
      v8u r;
      r.lo = *(const u32x4*)&sA[buf][idx];
      r.hi = *(const u32x4*)&sA[buf][idx + 4];             // +16B
      af[i] = __builtin_bit_cast(v16bf, r);
    }
#pragma unroll
    for (int j = 0; j < 2; ++j) {
      const int idx = (wn + j * 16 + l15) * 16 + lhi * 8;  // bytes: col*64+lhi*32
      v8u r;
      r.lo = *(const u32x4*)&sB[buf][idx];
      r.hi = *(const u32x4*)&sB[buf][idx + 4];
      bfg[j] = __builtin_bit_cast(v16bf, r);
    }
#pragma unroll
    for (int i = 0; i < 4; ++i)
#pragma unroll
      for (int j = 0; j < 2; ++j)
        acc[i][j] = __builtin_amdgcn_wmma_f32_16x16x32_bf16(
            false, af[i], false, bfg[j], (short)0, acc[i][j], false, false);
  };

  // prologue: tile 0 -> buffer 0
  stage_a(0, 0);
  load_b(0);
  store_b(0);
  wait_async0();
  __syncthreads();

  const int nT = K >> 5;
  for (int it = 0; it < nT; ++it) {
    const int cur = it & 1, nxt = cur ^ 1;
    const bool pf = (it + 1 < nT);
    if (pf) {                       // prefetch tile it+1 while computing it
      stage_a((it + 1) << 5, nxt);  // async engine writes LDS behind compute
      load_b((it + 1) << 5);
    }
    compute(cur);
    if (pf) store_b(nxt);
    wait_async0();
    __syncthreads();
  }

  // Epilogue. C/D layout: VGPR r, lanes 0-15 -> M=r, N=lane; lanes 16-31 -> M=r+8.
#pragma unroll
  for (int j = 0; j < 2; ++j) {
    const int col = n0 + wn + j * 16 + l15;
    const float bcol = bia[col];
#pragma unroll
    for (int i = 0; i < 4; ++i) {
#pragma unroll
      for (int r = 0; r < 8; ++r) {
        const int row = m0 + wm + i * 16 + (lhi ? r + 8 : r);
        float v = acc[i][j][r] + bcol;
        if (GELU_BF16) {
          v = 0.5f * v * (1.0f + erff(v * 0.70710678118654752f));
          ((unsigned short*)Obase)[(size_t)e * oBatch + (size_t)row * Nn + col] = f2bf(v);
        } else {
          ((float*)Obase)[(size_t)e * oBatch + (size_t)row * Nn + col] = v;
        }
      }
    }
  }
}

// ---------------------------------------------------------------------------
// Combine: out[t] = w0*y[flat0] + w1*y[flat1]  (dropped tokens contribute 0).
// ---------------------------------------------------------------------------
__global__ void __launch_bounds__(256) k_combine(
    const float* __restrict__ y,
    const int* __restrict__ flat0, const int* __restrict__ flat1,
    const float* __restrict__ w0, const float* __restrict__ w1,
    float* __restrict__ out)
{
  const int t = blockIdx.x, tid = threadIdx.x;
  const int d = tid * 4;
  f32x4 r = f32x4{0.f, 0.f, 0.f, 0.f};
  const int f0 = flat0[t]; const float a0 = w0[t];
  if (f0 >= 0) { f32x4 v = *(const f32x4*)&y[(size_t)f0 * Dm + d]; r += v * a0; }
  const int f1 = flat1[t]; const float a1 = w1[t];
  if (f1 >= 0) { f32x4 v = *(const f32x4*)&y[(size_t)f1 * Dm + d]; r += v * a1; }
  *(f32x4*)&out[(size_t)t * Dm + d] = r;
}

// ---------------------------------------------------------------------------
extern "C" void kernel_launch(void* const* d_in, const int* in_sizes, int n_in,
                              void* d_out, int out_size, void* d_ws, size_t ws_size,
                              hipStream_t stream) {
  (void)in_sizes; (void)n_in; (void)out_size; (void)ws_size;
  const float* x     = (const float*)d_in[0];
  const float* wg    = (const float*)d_in[1];
  const float* fc1_w = (const float*)d_in[2];
  const float* fc1_b = (const float*)d_in[3];
  const float* fc2_w = (const float*)d_in[4];
  const float* fc2_b = (const float*)d_in[5];
  float* out = (float*)d_out;

  // workspace layout (256B aligned regions)
  char* ws = (char*)d_ws;
  size_t off = 0;
  auto take = [&](size_t bytes) -> char* {
    char* p = ws + off;
    off = (off + bytes + 255) & ~(size_t)255;
    return p;
  };
  float*          scores  = (float*)take((size_t)Ntok * Ex * 4);
  int*            eid0    = (int*)  take((size_t)Ntok * 4);
  int*            eid1    = (int*)  take((size_t)Ntok * 4);
  float*          gv0     = (float*)take((size_t)Ntok * 4);
  float*          gv1     = (float*)take((size_t)Ntok * 4);
  int*            flat0   = (int*)  take((size_t)Ntok * 4);
  int*            flat1   = (int*)  take((size_t)Ntok * 4);
  float*          w0      = (float*)take((size_t)Ntok * 4);
  float*          w1      = (float*)take((size_t)Ntok * 4);
  int*            counts0 = (int*)  take((size_t)Ex * 4);
  unsigned short* disp    = (unsigned short*)take((size_t)Ex * Cap * Dm * 2);  // 42 MB
  unsigned short* hbuf    = (unsigned short*)take((size_t)Ex * Cap * Hm * 2);  // 168 MB
  float*          ybuf    = (float*)take((size_t)Ex * Cap * Dm * 4);           // 84 MB

  (void)hipMemsetAsync(disp, 0, (size_t)Ex * Cap * Dm * 2, stream);
  (void)hipMemsetAsync(counts0, 0, (size_t)Ex * 4, stream);

  k_gate<<<Ntok / 8, 256, 0, stream>>>(x, wg, scores, eid0, eid1, gv0, gv1, counts0);
  k_laux<<<1, 256, 0, stream>>>(scores, counts0, out + (size_t)Ntok * Dm);
  k_rank<<<Ntok / 8, 256, 0, stream>>>(eid0, eid1, gv0, gv1, counts0,
                                       flat0, flat1, w0, w1);
  k_scatter<<<Ntok, 256, 0, stream>>>(x, flat0, flat1, disp);

  // GEMM1: (C x D) @ (D x H) + b1, GELU, bf16 out.  grid = (20*32, 8)
  k_gemm<1><<<dim3((Cap / 128) * (Hm / 128), Ex), 256, 0, stream>>>(
      disp, fc1_w, fc1_b, (void*)hbuf,
      Cap, Hm, Dm, Cap / 128,
      (long long)Cap * Dm, (long long)Dm * Hm, (long long)Hm, (long long)Cap * Hm);

  // GEMM2: (C x H) @ (H x D) + b2, f32 out.  grid = (20*8, 8)
  k_gemm<0><<<dim3((Cap / 128) * (Dm / 128), Ex), 256, 0, stream>>>(
      hbuf, fc2_w, fc2_b, (void*)ybuf,
      Cap, Dm, Hm, Cap / 128,
      (long long)Cap * Hm, (long long)Hm * Dm, (long long)Dm, (long long)Cap * Dm);

  k_combine<<<Ntok, 256, 0, stream>>>(ybuf, flat0, flat1, w0, w1, out);
}
